// MoE_15719580304362
// MI455X (gfx1250) — matrix operand order repaired and checked by reading, not compiled, not run on hardware
//
#include <hip/hip_runtime.h>
#include <math.h>

#define DIM  1024
#define HID  512
#define SHID 1024
#define E    8
#define TOPK 2

#define AS1 __attribute__((address_space(1)))
#define AS3 __attribute__((address_space(3)))

#if defined(__has_builtin)
#if __has_builtin(__builtin_amdgcn_global_load_async_to_lds_b128) && \
    __has_builtin(__builtin_amdgcn_s_wait_asynccnt)
#define USE_ASYNC 1
#endif
#endif
#ifndef USE_ASYNC
#define USE_ASYNC 0
#endif

typedef float v2f  __attribute__((ext_vector_type(2)));
typedef float v8f  __attribute__((ext_vector_type(8)));
typedef int   v4i  __attribute__((ext_vector_type(4)));
typedef __bf16 v8bf  __attribute__((ext_vector_type(8)));
typedef __bf16 v16bf __attribute__((ext_vector_type(16)));

__device__ __forceinline__ v8f wmma_f32_k4(v2f a, v2f b, v8f c) {
  return __builtin_amdgcn_wmma_f32_16x16x4_f32(false, a, false, b, (short)0, c,
                                               false, false);
}

__device__ __forceinline__ v8f wmma_bf16_k32(v16bf a, v16bf b, v8f c) {
  return __builtin_amdgcn_wmma_f32_16x16x32_bf16(false, a, false, b, (short)0, c,
                                                 false, false);
}

__device__ __forceinline__ v8bf ld8bf(const __bf16* p) {
  return *(const v8bf*)p;   // global_load_b128
}

__device__ __forceinline__ v16bf cat16(v8bf a, v8bf b) {
  return __builtin_shufflevector(a, b, 0, 1, 2, 3, 4, 5, 6, 7,
                                 8, 9, 10, 11, 12, 13, 14, 15);
}

__device__ __forceinline__ float silu(float v) {
  return v / (1.f + __expf(-v));
}

// stage 16 bytes global -> LDS (async if available)
__device__ __forceinline__ void stage16(const __bf16* g, __bf16* l) {
#if USE_ASYNC
  __builtin_amdgcn_global_load_async_to_lds_b128((AS1 v4i*)g, (AS3 v4i*)l, 0, 0);
#else
  *(v8bf*)l = *(const v8bf*)g;
#endif
}

__device__ __forceinline__ void wait_stage() {
#if USE_ASYNC
  __builtin_amdgcn_s_wait_asynccnt(0);
#endif
}

// ---------------- fp32 -> bf16 hi/lo split ----------------

__global__ void cvt_kernel(const float* __restrict__ src, __bf16* __restrict__ hi,
                           __bf16* __restrict__ lo, int n) {
  int i = blockIdx.x * blockDim.x + threadIdx.x;
  if (i < n) {
    float v = src[i];
    __bf16 h = (__bf16)v;
    hi[i] = h;
    lo[i] = (__bf16)(v - (float)h);
  }
}

// ---------------- routing ----------------

__global__ void init_kernel(int* counts) {
  if (threadIdx.x < E) counts[threadIdx.x] = 0;
}

__global__ void gate_kernel(const float* __restrict__ x,
                            const float* __restrict__ wg,
                            const float* __restrict__ bias,
                            int* __restrict__ tki, float* __restrict__ wts,
                            int* __restrict__ counts, int N) {
  int wave = threadIdx.x >> 5;
  int lane = threadIdx.x & 31;
  int n = blockIdx.x * (blockDim.x >> 5) + wave;
  if (n >= N) return;
  const float* xr = x + (size_t)n * DIM;
  float logit[E];
#pragma unroll
  for (int e = 0; e < E; ++e) {
    const float* wr = wg + e * DIM;
    float p = 0.f;
    for (int d = lane; d < DIM; d += 32) p += xr[d] * wr[d];
#pragma unroll
    for (int off = 16; off > 0; off >>= 1) p += __shfl_xor(p, off, 32);
    logit[e] = p;
  }
  if (lane == 0) {
    float m = logit[0];
#pragma unroll
    for (int e = 1; e < E; ++e) m = fmaxf(m, logit[e]);
    float sc[E]; float sum = 0.f;
#pragma unroll
    for (int e = 0; e < E; ++e) { sc[e] = __expf(logit[e] - m); sum += sc[e]; }
    float inv = 1.f / sum;
    float s[E];
#pragma unroll
    for (int e = 0; e < E; ++e) { sc[e] *= inv; s[e] = sc[e] + bias[e]; }
    float gs[4];
#pragma unroll
    for (int g = 0; g < 4; ++g) gs[g] = s[2 * g] + s[2 * g + 1];
    int g1 = 0;
#pragma unroll
    for (int g = 1; g < 4; ++g) if (gs[g] > gs[g1]) g1 = g;
    int g2 = (g1 == 0) ? 1 : 0;
#pragma unroll
    for (int g = 0; g < 4; ++g) if (g != g1 && gs[g] > gs[g2]) g2 = g;
    float msk[E];
#pragma unroll
    for (int e = 0; e < E; ++e) {
      int g = e >> 1;
      msk[e] = (g == g1 || g == g2) ? s[e] : -INFINITY;
    }
    int i1 = 0;
#pragma unroll
    for (int e = 1; e < E; ++e) if (msk[e] > msk[i1]) i1 = e;
    int i2 = (i1 == 0) ? 1 : 0;
#pragma unroll
    for (int e = 0; e < E; ++e) if (e != i1 && msk[e] > msk[i2]) i2 = e;
    tki[n * 2 + 0] = i1;
    tki[n * 2 + 1] = i2;
    wts[n * 2 + 0] = sc[i1];   // ROUTE_SCALE = 1.0
    wts[n * 2 + 1] = sc[i2];
    atomicAdd(&counts[i1], 1);
    atomicAdd(&counts[i2], 1);
  }
}

__global__ void offs_kernel(const int* counts, int* offs) {
  if (threadIdx.x == 0) {
    int o = 0;
    for (int e = 0; e < E; ++e) { offs[e] = o; o += counts[e]; }
    offs[E] = o;
  }
}

// ---------------- Y1/Y3 = X[0..7] @ w1[e], w3[e] (exact fp32 WMMA) ----------------

__global__ void y_kernel(const float* __restrict__ x,
                         const float* __restrict__ w1,
                         const float* __restrict__ w3,
                         float* __restrict__ y1, float* __restrict__ y3) {
  int e = blockIdx.x;
  int s0 = blockIdx.y * 16;
  int lane = threadIdx.x & 31;
  int m = lane & 15;
  int half = lane >> 4;
  const float* w1e = w1 + (size_t)e * DIM * HID;
  const float* w3e = w3 + (size_t)e * DIM * HID;
  v8f acc1 = {}; v8f acc3 = {};
  for (int k = 0; k < DIM; k += 4) {
    int kk = k + 2 * half;
    v2f a;
    if (m < 8) { const float* p = x + (size_t)m * DIM + kk; a[0] = p[0]; a[1] = p[1]; }
    else       { a[0] = 0.f; a[1] = 0.f; }
    v2f b1, b3;
    b1[0] = w1e[(size_t)kk * HID + s0 + m];
    b1[1] = w1e[(size_t)(kk + 1) * HID + s0 + m];
    b3[0] = w3e[(size_t)kk * HID + s0 + m];
    b3[1] = w3e[(size_t)(kk + 1) * HID + s0 + m];
    acc1 = wmma_f32_k4(a, b1, acc1);
    acc3 = wmma_f32_k4(a, b3, acc3);
  }
  if (half == 0) {
#pragma unroll
    for (int i = 0; i < 8; ++i) {
      y1[((size_t)i * E + e) * HID + s0 + m] = acc1[i];
      y3[((size_t)i * E + e) * HID + s0 + m] = acc3[i];
    }
  }
}

// ---------------- hsum accumulation over slots ----------------

__global__ void hsum_kernel(const int* __restrict__ offs,
                            const int* __restrict__ tki,
                            const float* __restrict__ wts,
                            const float* __restrict__ y1,
                            const float* __restrict__ y3,
                            float* __restrict__ hsum) {
  int e = blockIdx.x;
  __shared__ float acc[E][HID];   // 16 KB
  for (int i = threadIdx.x; i < E * HID; i += blockDim.x)
    ((float*)acc)[i] = 0.f;
  __syncthreads();
  int r0 = offs[e], r1 = offs[e + 1];
  for (int r = r0; r < r1; ++r) {
    int f = tki[r];
    float w = wts[r];
    const float* y1p = y1 + ((size_t)f * E + e) * HID;
    const float* y3p = y3 + ((size_t)f * E + e) * HID;
    for (int s = threadIdx.x; s < HID; s += blockDim.x) {
      float a = w * y1p[s];
      float b = w * y3p[s];
      acc[f][s] += silu(a) * b;   // race-free: column s owned by this thread
    }
  }
  __syncthreads();
  for (int f = 0; f < E; ++f)
    for (int s = threadIdx.x; s < HID; s += blockDim.x)
      hsum[(size_t)f * (E * HID) + (size_t)e * HID + s] = acc[f][s];
}

// ---------------- shared FFN GEMM1 (split-bf16 WMMA, async B staging in LDS) ----------------
// block = 4 waves, 128 rows x 32 cols; B panels (s1h/s1l/s3h/s3l) staged once per block.

__global__ void mm1_kernel(const __bf16* __restrict__ xh, const __bf16* __restrict__ xl,
                           const __bf16* __restrict__ s1h, const __bf16* __restrict__ s1l,
                           const __bf16* __restrict__ s3h, const __bf16* __restrict__ s3l,
                           __bf16* __restrict__ acth, __bf16* __restrict__ actl, int N) {
  __shared__ __bf16 lb[2][4][32 * 32];   // double buffer x {s1h,s1l,s3h,s3l} x 2KB
  int tid = threadIdx.x;
  int wv = tid >> 5, lane = tid & 31;
  int rb = blockIdx.x * 128 + wv * 32;
  int cb = blockIdx.y * 32;
  int nn = lane & 15, half = lane >> 4;
  // staging role: thread t fills col = t/4, 8-element quarter q = t%4
  int scol = tid >> 2, sq = tid & 3;
  size_t gofB = (size_t)(cb + scol) * DIM + sq * 8;
  int lofB = scol * 32 + sq * 8;
  const __bf16* gB0 = s1h + gofB;
  const __bf16* gB1 = s1l + gofB;
  const __bf16* gB2 = s3h + gofB;
  const __bf16* gB3 = s3l + gofB;

  // prologue: fill buffer 0 with k0 = 0
  stage16(gB0, &lb[0][0][lofB]);
  stage16(gB1, &lb[0][1][lofB]);
  stage16(gB2, &lb[0][2][lofB]);
  stage16(gB3, &lb[0][3][lofB]);

  v8f acc1[2][2] = {}; v8f acc3[2][2] = {};
  for (int k0 = 0; k0 < DIM; k0 += 32) {
    int buf = (k0 >> 5) & 1;
    wait_stage();          // current buffer resident
    __syncthreads();       // everyone done reading the other buffer
    if (k0 + 32 < DIM) {   // prefetch next panel into other buffer
      stage16(gB0 + k0 + 32, &lb[buf ^ 1][0][lofB]);
      stage16(gB1 + k0 + 32, &lb[buf ^ 1][1][lofB]);
      stage16(gB2 + k0 + 32, &lb[buf ^ 1][2][lofB]);
      stage16(gB3 + k0 + 32, &lb[buf ^ 1][3][lofB]);
    }
    int aOff = k0 + 8 * half;      // A runs: K = 8h..8h+7 and 16+8h..+7
    v16bf ah[2], al[2];
#pragma unroll
    for (int mi = 0; mi < 2; ++mi) {
      size_t row = (size_t)(rb + 16 * mi + nn) * DIM;
      ah[mi] = cat16(ld8bf(xh + row + aOff), ld8bf(xh + row + aOff + 16));
      al[mi] = cat16(ld8bf(xl + row + aOff), ld8bf(xl + row + aOff + 16));
    }
    v16bf b1h[2], b1l[2], b3h[2], b3l[2];
#pragma unroll
    for (int ni = 0; ni < 2; ++ni) {
      int cc = (16 * ni + nn) * 32 + 8 * half;   // B runs: K = 8h..8h+7 and 16+8h..+7
      b1h[ni] = cat16(*(const v8bf*)&lb[buf][0][cc], *(const v8bf*)&lb[buf][0][cc + 16]);
      b1l[ni] = cat16(*(const v8bf*)&lb[buf][1][cc], *(const v8bf*)&lb[buf][1][cc + 16]);
      b3h[ni] = cat16(*(const v8bf*)&lb[buf][2][cc], *(const v8bf*)&lb[buf][2][cc + 16]);
      b3l[ni] = cat16(*(const v8bf*)&lb[buf][3][cc], *(const v8bf*)&lb[buf][3][cc + 16]);
    }
#pragma unroll
    for (int mi = 0; mi < 2; ++mi)
#pragma unroll
      for (int ni = 0; ni < 2; ++ni) {
        acc1[mi][ni] = wmma_bf16_k32(ah[mi], b1h[ni], acc1[mi][ni]);
        acc1[mi][ni] = wmma_bf16_k32(ah[mi], b1l[ni], acc1[mi][ni]);
        acc1[mi][ni] = wmma_bf16_k32(al[mi], b1h[ni], acc1[mi][ni]);
        acc3[mi][ni] = wmma_bf16_k32(ah[mi], b3h[ni], acc3[mi][ni]);
        acc3[mi][ni] = wmma_bf16_k32(ah[mi], b3l[ni], acc3[mi][ni]);
        acc3[mi][ni] = wmma_bf16_k32(al[mi], b3h[ni], acc3[mi][ni]);
      }
  }
#pragma unroll
  for (int mi = 0; mi < 2; ++mi)
#pragma unroll
    for (int ni = 0; ni < 2; ++ni)
#pragma unroll
      for (int i = 0; i < 8; ++i) {
        int row = rb + 16 * mi + i + 8 * half;
        int col = cb + 16 * ni + nn;
        float v = silu(acc1[mi][ni][i]) * acc3[mi][ni][i];
        __bf16 h = (__bf16)v;
        acth[(size_t)row * SHID + col] = h;
        actl[(size_t)row * SHID + col] = (__bf16)(v - (float)h);
      }
}

// ---------------- shared FFN GEMM2 (split-bf16 WMMA): out = act @ sw2^T ----------------

__global__ void mm2_kernel(const __bf16* __restrict__ ath, const __bf16* __restrict__ atl,
                           const __bf16* __restrict__ s2h, const __bf16* __restrict__ s2l,
                           float* __restrict__ out, int N) {
  int rb = blockIdx.x * 32;
  int cb = blockIdx.y * 32;
  int lane = threadIdx.x & 31;
  int nn = lane & 15;
  int half = lane >> 4;
  v8f acc[2][2] = {};
  for (int k0 = 0; k0 < SHID; k0 += 32) {
    int aOff = k0 + 8 * half;
    int bOff = k0 + 8 * half;
    v16bf ah[2], al[2], bh[2], bl[2];
#pragma unroll
    for (int mi = 0; mi < 2; ++mi) {
      size_t row = (size_t)(rb + 16 * mi + nn) * SHID;
      ah[mi] = cat16(ld8bf(ath + row + aOff), ld8bf(ath + row + aOff + 16));
      al[mi] = cat16(ld8bf(atl + row + aOff), ld8bf(atl + row + aOff + 16));
    }
#pragma unroll
    for (int ni = 0; ni < 2; ++ni) {
      size_t col = (size_t)(cb + 16 * ni + nn) * SHID;
      bh[ni] = cat16(ld8bf(s2h + col + bOff), ld8bf(s2h + col + bOff + 16));
      bl[ni] = cat16(ld8bf(s2l + col + bOff), ld8bf(s2l + col + bOff + 16));
    }
#pragma unroll
    for (int mi = 0; mi < 2; ++mi)
#pragma unroll
      for (int ni = 0; ni < 2; ++ni) {
        acc[mi][ni] = wmma_bf16_k32(ah[mi], bh[ni], acc[mi][ni]);
        acc[mi][ni] = wmma_bf16_k32(ah[mi], bl[ni], acc[mi][ni]);
        acc[mi][ni] = wmma_bf16_k32(al[mi], bh[ni], acc[mi][ni]);
      }
  }
#pragma unroll
  for (int mi = 0; mi < 2; ++mi)
#pragma unroll
    for (int ni = 0; ni < 2; ++ni)
#pragma unroll
      for (int i = 0; i < 8; ++i) {
        int row = rb + 16 * mi + i + 8 * half;
        int col = cb + 16 * ni + nn;
        out[(size_t)row * DIM + col] = acc[mi][ni][i];
      }
}

// ---------------- out[0..7] += hsum(8 x 4096) @ w2_flat(4096 x 1024) (exact fp32) ----------------

__global__ void final_kernel(const float* __restrict__ hsum,
                             const float* __restrict__ w2,
                             float* __restrict__ out) {
  int cb = blockIdx.x * 16;
  int lane = threadIdx.x & 31;
  int nn = lane & 15;
  int half = lane >> 4;
  v8f acc = {};
  for (int k = 0; k < E * HID; k += 4) {
    int kk = k + 2 * half;
    v2f a, b;
    if (nn < 8) { const float* p = hsum + (size_t)nn * (E * HID) + kk; a[0] = p[0]; a[1] = p[1]; }
    else        { a[0] = 0.f; a[1] = 0.f; }
    b[0] = w2[(size_t)kk * DIM + cb + nn];
    b[1] = w2[(size_t)(kk + 1) * DIM + cb + nn];
    acc = wmma_f32_k4(a, b, acc);
  }
  if (half == 0) {
#pragma unroll
    for (int i = 0; i < 8; ++i)
      out[(size_t)i * DIM + cb + nn] += acc[i];
  }
}

// ---------------- launch ----------------

extern "C" void kernel_launch(void* const* d_in, const int* in_sizes, int n_in,
                              void* d_out, int out_size, void* d_ws, size_t ws_size,
                              hipStream_t stream) {
  const float* x    = (const float*)d_in[0];
  const float* wg   = (const float*)d_in[1];
  const float* w1   = (const float*)d_in[2];
  const float* w2   = (const float*)d_in[3];
  const float* w3   = (const float*)d_in[4];
  const float* sw1  = (const float*)d_in[5];
  const float* sw2  = (const float*)d_in[6];
  const float* sw3  = (const float*)d_in[7];
  const float* bias = (const float*)d_in[8];
  float* out = (float*)d_out;
  int N = in_sizes[0] / DIM;   // 4096

  char* ws = (char*)d_ws;
  int*   counts = (int*)(ws + 0);
  int*   offs   = (int*)(ws + 128);
  int*   tki    = (int*)(ws + 256);
  float* wts    = (float*)(ws + 256 + (size_t)N * TOPK * 4);
  char*  p      = ws + 256 + (size_t)N * TOPK * 8;
  float* y1   = (float*)p;  p += (size_t)E * E * HID * 4;
  float* y3   = (float*)p;  p += (size_t)E * E * HID * 4;
  float* hsum = (float*)p;  p += (size_t)E * E * HID * 4;
  __bf16* xh  = (__bf16*)p; p += (size_t)N * DIM * 2;
  __bf16* xl  = (__bf16*)p; p += (size_t)N * DIM * 2;
  __bf16* s1h = (__bf16*)p; p += (size_t)SHID * DIM * 2;
  __bf16* s1l = (__bf16*)p; p += (size_t)SHID * DIM * 2;
  __bf16* s3h = (__bf16*)p; p += (size_t)SHID * DIM * 2;
  __bf16* s3l = (__bf16*)p; p += (size_t)SHID * DIM * 2;
  __bf16* s2h = (__bf16*)p; p += (size_t)DIM * SHID * 2;
  __bf16* s2l = (__bf16*)p; p += (size_t)DIM * SHID * 2;
  __bf16* ath = (__bf16*)p; p += (size_t)N * SHID * 2;
  __bf16* atl = (__bf16*)p;

  // routing chain
  init_kernel<<<1, 32, 0, stream>>>(counts);
  gate_kernel<<<dim3((N + 7) / 8), dim3(256), 0, stream>>>(x, wg, bias, tki, wts, counts, N);
  offs_kernel<<<1, 32, 0, stream>>>(counts, offs);
  y_kernel<<<dim3(E, HID / 16), dim3(32), 0, stream>>>(x, w1, w3, y1, y3);
  hsum_kernel<<<dim3(E), dim3(256), 0, stream>>>(offs, tki, wts, y1, y3, hsum);

  // bf16 hi/lo splits
  int nx = N * DIM, nw = SHID * DIM;
  cvt_kernel<<<dim3((nx + 255) / 256), dim3(256), 0, stream>>>(x, xh, xl, nx);
  cvt_kernel<<<dim3((nw + 255) / 256), dim3(256), 0, stream>>>(sw1, s1h, s1l, nw);
  cvt_kernel<<<dim3((nw + 255) / 256), dim3(256), 0, stream>>>(sw3, s3h, s3l, nw);
  cvt_kernel<<<dim3((nw + 255) / 256), dim3(256), 0, stream>>>(sw2, s2h, s2l, nw);

  // shared-expert FFN on the bf16 matrix pipe
  mm1_kernel<<<dim3(N / 128, SHID / 32), dim3(128), 0, stream>>>(xh, xl, s1h, s1l, s3h, s3l, ath, atl, N);
  mm2_kernel<<<dim3(N / 32, DIM / 32), dim3(32), 0, stream>>>(ath, atl, s2h, s2l, out, N);

  // routed contribution into rows 0..7
  final_kernel<<<dim3(DIM / 16), dim3(32), 0, stream>>>(hsum, w2, out);
}
